// ActorGAT_72138270703869
// MI455X (gfx1250) — matrix operand order repaired and checked by reading, not compiled, not run on hardware
//
#include <hip/hip_runtime.h>

// ---------------------------------------------------------------------------
// Types for CDNA5 WMMA (wave32)
// ---------------------------------------------------------------------------
typedef __attribute__((ext_vector_type(16))) __bf16 v16bf;
typedef __attribute__((ext_vector_type(8)))  __bf16 v8bf;
typedef __attribute__((ext_vector_type(8)))  float  v8f;
typedef __attribute__((ext_vector_type(4)))  int    v4i;

#if __has_builtin(__builtin_amdgcn_global_load_async_to_lds_b128) && \
    __has_builtin(__builtin_amdgcn_s_wait_asynccnt)
#define USE_ASYNC 1
typedef __attribute__((address_space(1))) v4i* gv4i_p;   // global int4*
typedef __attribute__((address_space(3))) v4i* lv4i_p;   // LDS int4*
#else
#define USE_ASYNC 0
#endif

__device__ __forceinline__ __bf16 f2bf(float f) {
  unsigned u = __float_as_uint(f);
  unsigned r = (u + 0x7FFFu + ((u >> 16) & 1u)) >> 16;   // round-to-nearest-even
  unsigned short hs = (unsigned short)r;
  __bf16 b;
  __builtin_memcpy(&b, &hs, 2);
  return b;
}

// Order-preserving float<->uint mapping for atomicMax on floats
__device__ __forceinline__ unsigned enc_f(float f) {
  unsigned u = __float_as_uint(f);
  return (u & 0x80000000u) ? ~u : (u | 0x80000000u);
}
__device__ __forceinline__ float dec_f(unsigned u) {
  return __uint_as_float((u & 0x80000000u) ? (u & 0x7FFFFFFFu) : ~u);
}
#define ENC_NEG_INF 0x007FFFFFu

__device__ __forceinline__ float sigm(float x) { return 1.0f / (1.0f + __expf(-x)); }

// ---------------------------------------------------------------------------
// lin0: out[n,c] = relu(x[n,0:3] @ W[3,256] + b)
// ---------------------------------------------------------------------------
__global__ __launch_bounds__(256) void lin0_k(const float* __restrict__ x,
                                              const float* __restrict__ W,
                                              const float* __restrict__ b,
                                              float* __restrict__ out) {
  int idx = blockIdx.x * 256 + threadIdx.x;   // N*256
  int n = idx >> 8, c = idx & 255;
  const float* xr = x + n * 3;
  float acc = b[c] + xr[0] * W[c] + xr[1] * W[256 + c] + xr[2] * W[512 + c];
  out[idx] = acc > 0.f ? acc : 0.f;
}

// ---------------------------------------------------------------------------
// f32 -> bf16 conversion (8 elements per thread, 16B vector store)
// ---------------------------------------------------------------------------
__global__ __launch_bounds__(256) void f32_to_bf16_k(const float* __restrict__ src,
                                                     __bf16* __restrict__ dst) {
  int i8 = (blockIdx.x * 256 + threadIdx.x) * 8;
  const float4* s = (const float4*)(src + i8);
  float4 a = s[0], b = s[1];
  v8bf o;
  o[0] = f2bf(a.x); o[1] = f2bf(a.y); o[2] = f2bf(a.z); o[3] = f2bf(a.w);
  o[4] = f2bf(b.x); o[5] = f2bf(b.y); o[6] = f2bf(b.z); o[7] = f2bf(b.w);
  *(v8bf*)(dst + i8) = o;
}

// W[K,Nout] f32 -> WT[Nout,K] bf16 (transpose; small, weights are L2 resident)
__global__ __launch_bounds__(256) void wt_bf16_k(const float* __restrict__ W,
                                                 __bf16* __restrict__ WT,
                                                 int K, int Nout) {
  int idx = blockIdx.x * 256 + threadIdx.x;   // K*Nout
  int n = idx / K, k = idx - n * K;
  WT[idx] = f2bf(W[(size_t)k * Nout + n]);
}

// ---------------------------------------------------------------------------
// bf16 WMMA GEMM: C[M,Nout] = A[M,K] x WT[Nout,K]^T, f32 accumulation.
// 128x128 block tile, 8 waves (4x2), 32x64 per wave.  Double-buffered LDS,
// staged with GLOBAL_LOAD_ASYNC_TO_LDS_B128 (ASYNCcnt) when available.
// ---------------------------------------------------------------------------
__global__ __launch_bounds__(256) void gemm_bf16_wmma(const __bf16* __restrict__ A,
                                                      const __bf16* __restrict__ WT,
                                                      float* __restrict__ C,
                                                      int K, int Nout) {
  __shared__ __align__(16) __bf16 Als[2][128][40];   // [buf][m][k], 80B row stride
  __shared__ __align__(16) __bf16 Bls[2][128][40];   // [buf][n][k]

  const int tid  = threadIdx.x;
  const int lane = tid & 31;
  const int wv   = tid >> 5;
  const int mBase = blockIdx.y * 128;
  const int nBase = blockIdx.x * 128;
  const int wm = wv >> 1, wn = wv & 1;            // 4 waves in M, 2 in N
  const int mOff = wm * 32, nOff = wn * 64;
  const int halfL = lane >> 4;
  const int lidx  = lane & 15;

  v8f acc[2][4];
  v8f zero = {0.f, 0.f, 0.f, 0.f, 0.f, 0.f, 0.f, 0.f};
#pragma unroll
  for (int i = 0; i < 2; ++i)
#pragma unroll
    for (int j = 0; j < 4; ++j) acc[i][j] = zero;

  // one 16B chunk per thread per matrix per tile
  const int cRow = tid >> 1, cSeg = tid & 1;
  const __bf16* aSrc = A  + (size_t)(mBase + cRow) * K + cSeg * 8;
  const __bf16* bSrc = WT + (size_t)(nBase + cRow) * K + cSeg * 8;

  const int KT = K >> 5;

  // prologue: stage tile 0 into buffer 0
#if USE_ASYNC
  __builtin_amdgcn_global_load_async_to_lds_b128(
      (gv4i_p)(aSrc), (lv4i_p)&Als[0][cRow][cSeg * 8], 0, 0);
  __builtin_amdgcn_global_load_async_to_lds_b128(
      (gv4i_p)(bSrc), (lv4i_p)&Bls[0][cRow][cSeg * 8], 0, 0);
#else
  *(v8bf*)&Als[0][cRow][cSeg * 8] = *(const v8bf*)(aSrc);
  *(v8bf*)&Bls[0][cRow][cSeg * 8] = *(const v8bf*)(bSrc);
#endif

  for (int kt = 0; kt < KT; ++kt) {
    // stage next tile into the other buffer, then wait for current tile
    if (kt + 1 < KT) {
      int nb = (kt + 1) & 1, k1 = (kt + 1) << 5;
#if USE_ASYNC
      __builtin_amdgcn_global_load_async_to_lds_b128(
          (gv4i_p)(aSrc + k1), (lv4i_p)&Als[nb][cRow][cSeg * 8], 0, 0);
      __builtin_amdgcn_global_load_async_to_lds_b128(
          (gv4i_p)(bSrc + k1), (lv4i_p)&Bls[nb][cRow][cSeg * 8], 0, 0);
      __builtin_amdgcn_s_wait_asynccnt(2);   // in-order: current tile landed
#else
      *(v8bf*)&Als[nb][cRow][cSeg * 8] = *(const v8bf*)(aSrc + k1);
      *(v8bf*)&Bls[nb][cRow][cSeg * 8] = *(const v8bf*)(bSrc + k1);
#endif
    }
#if USE_ASYNC
    else {
      __builtin_amdgcn_s_wait_asynccnt(0);
    }
#endif
    __syncthreads();

    const int cb = kt & 1;
    // ---- fragments per ISA 16-bit layouts ----
    v16bf afr[2], bfr[4];
#pragma unroll
    for (int ms = 0; ms < 2; ++ms) {
      int row = mOff + ms * 16 + lidx;
      ((v8bf*)&afr[ms])[0] = *(const v8bf*)&Als[cb][row][halfL * 8];       // K 0..7 / 8..15
      ((v8bf*)&afr[ms])[1] = *(const v8bf*)&Als[cb][row][16 + halfL * 8];  // K 16..23 / 24..31
    }
#pragma unroll
    for (int ns = 0; ns < 4; ++ns) {
      int col = nOff + ns * 16 + lidx;
      ((v8bf*)&bfr[ns])[0] = *(const v8bf*)&Bls[cb][col][halfL * 8];       // K 0..7 / 8..15
      ((v8bf*)&bfr[ns])[1] = *(const v8bf*)&Bls[cb][col][16 + halfL * 8];  // K 16..23 / 24..31
    }
#pragma unroll
    for (int ms = 0; ms < 2; ++ms)
#pragma unroll
      for (int ns = 0; ns < 4; ++ns)
        acc[ms][ns] = __builtin_amdgcn_wmma_f32_16x16x32_bf16(
            false, afr[ms], false, bfr[ns], (short)0, acc[ms][ns], false, false);
    __syncthreads();
  }

  // ---- epilogue: C/D layout — VGPR r holds M = r + 8*halfL, N = lane&15 ----
#pragma unroll
  for (int ms = 0; ms < 2; ++ms)
#pragma unroll
    for (int ns = 0; ns < 4; ++ns) {
      int m0 = mBase + mOff + ms * 16 + 8 * halfL;
      int c0 = nBase + nOff + ns * 16 + lidx;
#pragma unroll
      for (int r = 0; r < 8; ++r)
        C[(size_t)(m0 + r) * Nout + c0] = acc[ms][ns][r];
    }
}

// ---------------------------------------------------------------------------
// GAT attention coefficients: es/ed per (node, head); also inits max/sum bufs
// ---------------------------------------------------------------------------
__global__ __launch_bounds__(256) void gat_coef(const float* __restrict__ xp,
                                                const float* __restrict__ a_s,
                                                const float* __restrict__ a_d,
                                                float* __restrict__ es,
                                                float* __restrict__ ed,
                                                unsigned* __restrict__ mu,
                                                float* __restrict__ sb, int H) {
  int wv = threadIdx.x >> 5, lane = threadIdx.x & 31;
  int p = blockIdx.x * 8 + wv;            // p in [0, N*H)
  int n = p / H, h = p - n * H;
  const float* xr  = xp + (size_t)n * H * 256 + h * 256;
  const float* asr = a_s + h * 256;
  const float* adr = a_d + h * 256;
  float s1 = 0.f, s2 = 0.f;
#pragma unroll
  for (int j = lane; j < 256; j += 32) { float v = xr[j]; s1 += v * asr[j]; s2 += v * adr[j]; }
#pragma unroll
  for (int o = 16; o; o >>= 1) { s1 += __shfl_xor(s1, o, 32); s2 += __shfl_xor(s2, o, 32); }
  if (lane == 0) { es[p] = s1; ed[p] = s2; mu[p] = ENC_NEG_INF; sb[p] = 0.f; }
}

// Pass 1: leaky-relu edge scores + per-dst segment max (float via uint atomicMax)
__global__ __launch_bounds__(256) void gat_edge_max(const int* __restrict__ ei, int E, int H,
                                                    const float* __restrict__ es,
                                                    const float* __restrict__ ed,
                                                    float* __restrict__ val,
                                                    unsigned* __restrict__ mu) {
  int idx = blockIdx.x * 256 + threadIdx.x;   // Etot*H
  int e = idx / H, h = idx - e * H;
  int s, d;
  if (e < E) { s = ei[e]; d = ei[E + e]; } else { s = d = e - E; }
  float v = es[s * H + h] + ed[d * H + h];
  v = v > 0.f ? v : 0.2f * v;
  val[idx] = v;
  atomicMax(&mu[d * H + h], enc_f(v));
}

// Pass 2: exp(v - max) and per-dst segment sum
__global__ __launch_bounds__(256) void gat_edge_exp(const int* __restrict__ ei, int E, int H,
                                                    float* __restrict__ val,
                                                    const unsigned* __restrict__ mu,
                                                    float* __restrict__ sb) {
  int idx = blockIdx.x * 256 + threadIdx.x;
  int e = idx / H, h = idx - e * H;
  int d = (e < E) ? ei[E + e] : (e - E);
  float ex = __expf(val[idx] - dec_f(mu[d * H + h]));
  val[idx] = ex;
  atomicAdd(&sb[d * H + h], ex);
}

// init aggregation output with broadcast bias
__global__ __launch_bounds__(256) void bias_init(float* __restrict__ out,
                                                 const float* __restrict__ b, int mask) {
  int idx = blockIdx.x * 256 + threadIdx.x;
  out[idx] = b[idx & mask];
}

// Pass 3: out[dst, hc] += (ex/s) * xp[src, hc]  (channel-parallel scatter-add)
__global__ __launch_bounds__(256) void gat_aggregate(const int* __restrict__ ei, int E,
                                                     int H, int HD,
                                                     const float* __restrict__ val,
                                                     const float* __restrict__ sb,
                                                     const float* __restrict__ xp,
                                                     float* __restrict__ out) {
  unsigned idx = blockIdx.x * 256u + threadIdx.x;   // Etot*HD
  int e = idx / HD, hc = idx - (unsigned)e * HD;
  int h = hc >> 8;
  int s, d;
  if (e < E) { s = ei[e]; d = ei[E + e]; } else { s = d = e - E; }
  float a = val[e * H + h] / sb[d * H + h];
  atomicAdd(&out[(size_t)d * HD + hc], a * xp[(size_t)s * HD + hc]);
}

// ---------------------------------------------------------------------------
// Zero init
// ---------------------------------------------------------------------------
__global__ __launch_bounds__(256) void zero_k(float* __restrict__ p) {
  p[blockIdx.x * 256 + threadIdx.x] = 0.f;
}

// ---------------------------------------------------------------------------
// LSTM cell: one block per batch row (B=64), 256 threads (one per hidden dim)
// ---------------------------------------------------------------------------
__global__ __launch_bounds__(256) void lstm_cell(const float* __restrict__ qs,
                                                 const float* __restrict__ hin,
                                                 const float* __restrict__ cin,
                                                 const float* __restrict__ Wih,
                                                 const float* __restrict__ Whh,
                                                 const float* __restrict__ bih,
                                                 const float* __restrict__ bhh,
                                                 float* __restrict__ hout,
                                                 float* __restrict__ cout) {
  __shared__ float q[512];
  __shared__ float hh[256];
  int g = blockIdx.x, tid = threadIdx.x;
  q[tid] = qs[g * 512 + tid];
  q[tid + 256] = qs[g * 512 + 256 + tid];
  hh[tid] = hin[g * 256 + tid];
  __syncthreads();
  float gate[4];
#pragma unroll
  for (int gi = 0; gi < 4; ++gi) {
    int r = gi * 256 + tid;
    float acc = bih[r] + bhh[r];
    const float* wi = Wih + (size_t)r * 512;
#pragma unroll 4
    for (int k = 0; k < 512; ++k) acc += q[k] * wi[k];
    const float* wh = Whh + (size_t)r * 256;
#pragma unroll 4
    for (int k = 0; k < 256; ++k) acc += hh[k] * wh[k];
    gate[gi] = acc;
  }
  float i_ = sigm(gate[0]), f_ = sigm(gate[1]);
  float gg = tanhf(gate[2]), o_ = sigm(gate[3]);
  float cn = f_ * cin[g * 256 + tid] + i_ * gg;
  cout[g * 256 + tid] = cn;
  hout[g * 256 + tid] = o_ * tanhf(cn);
}

// ---------------------------------------------------------------------------
// Set2Set attention step: graphs are contiguous (512 nodes each)
// ---------------------------------------------------------------------------
__global__ __launch_bounds__(256) void s2s_attn(const float* __restrict__ nodeF,
                                                const float* __restrict__ h,
                                                float* __restrict__ q_star) {
  __shared__ float hq[256];
  __shared__ float ev[512];
  __shared__ float red[256];
  int g = blockIdx.x, tid = threadIdx.x, lane = tid & 31, wv = tid >> 5;
  hq[tid] = h[g * 256 + tid];
  __syncthreads();
  for (int nn = wv; nn < 512; nn += 8) {
    const float* xr = nodeF + (size_t)(g * 512 + nn) * 256;
    float s = 0.f;
    for (int j = lane; j < 256; j += 32) s += xr[j] * hq[j];
#pragma unroll
    for (int o = 16; o; o >>= 1) s += __shfl_xor(s, o, 32);
    if (!lane) ev[nn] = s;
  }
  __syncthreads();
  red[tid] = fmaxf(ev[tid], ev[tid + 256]);
  __syncthreads();
  for (int st = 128; st; st >>= 1) { if (tid < st) red[tid] = fmaxf(red[tid], red[tid + st]); __syncthreads(); }
  float M = red[0];
  __syncthreads();
  float e0 = __expf(ev[tid] - M), e1 = __expf(ev[tid + 256] - M);
  red[tid] = e0 + e1;
  __syncthreads();
  for (int st = 128; st; st >>= 1) { if (tid < st) red[tid] += red[tid + st]; __syncthreads(); }
  float S = red[0];
  __syncthreads();
  ev[tid] = e0 / S;
  ev[tid + 256] = e1 / S;
  __syncthreads();
  float acc = 0.f;
  for (int nn = 0; nn < 512; ++nn) acc += ev[nn] * nodeF[(size_t)(g * 512 + nn) * 256 + tid];
  q_star[g * 512 + tid] = hq[tid];
  q_star[g * 512 + 256 + tid] = acc;
}

// ---------------------------------------------------------------------------
// Final MLP: z = [h1[nrbidx[t]], out[nonring[t,0..3]]] -> relu lin1 -> lin2
// ---------------------------------------------------------------------------
__global__ __launch_bounds__(256) void final_mlp(const float* __restrict__ nodeF,
                                                 const float* __restrict__ h1,
                                                 const int* __restrict__ nonring,
                                                 const int* __restrict__ nrbidx,
                                                 const float* __restrict__ W1,
                                                 const float* __restrict__ b1,
                                                 const float* __restrict__ W2,
                                                 const float* __restrict__ b2,
                                                 float* __restrict__ logits) {
  __shared__ float z[1280];
  __shared__ float z2[256];
  int t = blockIdx.x, tid = threadIdx.x;
  z[tid] = h1[nrbidx[t] * 256 + tid];
#pragma unroll
  for (int q = 0; q < 4; ++q) {
    int node = nonring[t * 4 + q];
    z[256 + q * 256 + tid] = nodeF[(size_t)node * 256 + tid];
  }
  __syncthreads();
  float acc = b1[tid];
#pragma unroll 4
  for (int k = 0; k < 1280; ++k) acc += z[k] * W1[k * 256 + tid];
  z2[tid] = acc > 0.f ? acc : 0.f;
  __syncthreads();
  if (tid < 6) {
    float a2 = b2[tid];
    for (int k = 0; k < 256; ++k) a2 += z2[k] * W2[k * 6 + tid];
    logits[t * 6 + tid] = a2;
  }
}

// ---------------------------------------------------------------------------
// Launch
// ---------------------------------------------------------------------------
extern "C" void kernel_launch(void* const* d_in, const int* in_sizes, int n_in,
                              void* d_out, int out_size, void* d_ws, size_t ws_size,
                              hipStream_t stream) {
  (void)in_sizes; (void)n_in; (void)out_size; (void)ws_size;
  const int N = 32768, Bg = 64, E = 262144, TPG = 48, T = Bg * TPG;
  const int Etot = E + N;

  const float* x      = (const float*)d_in[0];
  const int*   ei     = (const int*)d_in[1];
  const int*   nonr   = (const int*)d_in[3];
  const int*   nrbidx = (const int*)d_in[4];
  const float* lin0W  = (const float*)d_in[6];
  const float* lin0b  = (const float*)d_in[7];
  const float *gatW[6], *asrc[6], *adst[6], *gbias[6];
  for (int i = 0; i < 6; ++i) {
    gatW[i]  = (const float*)d_in[8 + i];
    asrc[i]  = (const float*)d_in[14 + i];
    adst[i]  = (const float*)d_in[20 + i];
    gbias[i] = (const float*)d_in[26 + i];
  }
  const float* s2sWih = (const float*)d_in[32];
  const float* s2sWhh = (const float*)d_in[33];
  const float* s2sbih = (const float*)d_in[34];
  const float* s2sbhh = (const float*)d_in[35];
  const float* memWih = (const float*)d_in[36];
  const float* memWhh = (const float*)d_in[37];
  const float* membih = (const float*)d_in[38];
  const float* membhh = (const float*)d_in[39];
  const float* lin1W  = (const float*)d_in[40];
  const float* lin1b  = (const float*)d_in[41];
  const float* lin2W  = (const float*)d_in[42];
  const float* lin2b  = (const float*)d_in[43];
  float* out = (float*)d_out;

  // workspace carve (ws poisoned with 0xAA; everything read is written first)
  char* w = (char*)d_ws;
  float*  bufA  = (float*)w;    w += (size_t)N * 512 * 4;   // activations / agg out
  float*  bufXP = (float*)w;    w += (size_t)N * 512 * 4;   // GEMM output xp
  __bf16* Abf   = (__bf16*)w;   w += (size_t)N * 512 * 2;   // bf16 activations
  __bf16* WTbf  = (__bf16*)w;   w += (size_t)512 * 512 * 2; // bf16 transposed weight
  float* es    = (float*)w;    w += (size_t)N * 2 * 4;
  float* ed    = (float*)w;    w += (size_t)N * 2 * 4;
  unsigned* mu = (unsigned*)w; w += (size_t)N * 2 * 4;
  float* sb    = (float*)w;    w += (size_t)N * 2 * 4;
  float* val   = (float*)w;    w += (size_t)Etot * 2 * 4;
  float* qstar = (float*)w;    w += (size_t)Bg * 512 * 4;   // qstar,hb,cb,z0 contiguous
  float* hb    = (float*)w;    w += (size_t)Bg * 256 * 4;
  float* cb    = (float*)w;    w += (size_t)Bg * 256 * 4;
  float* z0    = (float*)w;    w += (size_t)Bg * 256 * 4;

  // lin0
  lin0_k<<<(N * 256) / 256, 256, 0, stream>>>(x, lin0W, lin0b, bufA);

  // 6 GAT layers
  const int Ks[6]  = {256, 512, 512, 512, 512, 512};
  const int HDs[6] = {512, 512, 512, 512, 512, 256};
  const int Hs[6]  = {2, 2, 2, 2, 2, 1};
  for (int L = 0; L < 6; ++L) {
    int K = Ks[L], HD = HDs[L], H = Hs[L];
    // convert activations + transposed weight to bf16
    f32_to_bf16_k<<<(N * K / 8) / 256, 256, 0, stream>>>(bufA, Abf);
    wt_bf16_k<<<(K * HD) / 256, 256, 0, stream>>>(gatW[L], WTbf, K, HD);
    dim3 gg(HD / 128, N / 128);
    gemm_bf16_wmma<<<gg, 256, 0, stream>>>(Abf, WTbf, bufXP, K, HD);
    gat_coef<<<(N * H) / 8, 256, 0, stream>>>(bufXP, asrc[L], adst[L], es, ed, mu, sb, H);
    gat_edge_max<<<(Etot * H) / 256, 256, 0, stream>>>(ei, E, H, es, ed, val, mu);
    gat_edge_exp<<<(Etot * H) / 256, 256, 0, stream>>>(ei, E, H, val, mu, sb);
    bias_init<<<(N * HD) / 256, 256, 0, stream>>>(bufA, gbias[L], HD - 1);
    gat_aggregate<<<(unsigned)(((size_t)Etot * HD) / 256), 256, 0, stream>>>(
        ei, E, H, HD, val, sb, bufXP, bufA);
  }

  // Set2Set (zero qstar/hb/cb/z0, then 6 steps)
  zero_k<<<(Bg * 1280) / 256, 256, 0, stream>>>(qstar);
  for (int it = 0; it < 6; ++it) {
    lstm_cell<<<Bg, 256, 0, stream>>>(qstar, hb, cb, s2sWih, s2sWhh, s2sbih, s2sbhh, hb, cb);
    s2s_attn<<<Bg, 256, 0, stream>>>(bufA, hb, qstar);
  }

  // memory LSTM, one step, zero state; write h1/c1 straight into d_out tail
  float* h1 = out + (size_t)T * 6;
  float* c1 = h1 + (size_t)Bg * 256;
  lstm_cell<<<Bg, 256, 0, stream>>>(qstar, z0, z0, memWih, memWhh, membih, membhh, h1, c1);

  // final MLP -> logits
  final_mlp<<<T, 256, 0, stream>>>(bufA, h1, nonr, nrbidx, lin1W, lin1b, lin2W, lin2b, out);
}
